// CaMoE_System_40072044871826
// MI455X (gfx1250) — matrix operand order repaired and checked by reading, not compiled, not run on hardware
//
#include <hip/hip_runtime.h>
#include <cstdint>
#include <cstddef>

// Problem constants (from reference)
#define B_  2
#define T_  2048
#define C_  1024
#define H_  16
#define HS_ 64
#define E_  8
#define NR_ 6
#define NT_ 2
#define P_  4
#define FF_ 4096
#define BT_ (B_*T_)

typedef __attribute__((ext_vector_type(16))) __bf16         v16bf;
typedef __attribute__((ext_vector_type(16))) unsigned short v16us;
typedef __attribute__((ext_vector_type(8)))  unsigned short v8us;
typedef __attribute__((ext_vector_type(8)))  float          v8f;

union FragU { v16us u; v16bf b; v8us h[2]; };

__device__ __forceinline__ unsigned short f2bf(float f) {
  unsigned u = __builtin_bit_cast(unsigned, f);
  u += 0x7FFFu + ((u >> 16) & 1u);            // round-to-nearest-even
  return (unsigned short)(u >> 16);
}
__device__ __forceinline__ v8us pack8(const float* __restrict__ s) {
  v8us r;
  #pragma unroll
  for (int i = 0; i < 8; ++i) r[i] = f2bf(s[i]);
  return r;
}
__device__ __forceinline__ float sigm(float x) { return 1.0f / (1.0f + __expf(-x)); }

// ---------------------------------------------------------------------------
// bf16 WMMA GEMM: C(MxN) = epilogue(A(MxK) * B(KxN)), fp32 in/out.
// Block tile 128x64, K-step 32, 256 threads = 8 waves.
// Wave (mi = wid&3, ng = wid>>2) computes a 32x32 patch: m-tiles {mi, mi+4},
// n-tiles {2ng, 2ng+1} -> 4 WMMAs per K-step from 2 A-frags x 2 B-frags.
//
// Double-buffered LDS + register staging: one barrier per K-step; the next
// tile's global loads + bf16 conversion overlap the current tile's WMMAs.
//
// LDS layouts are "fragment ready":
//   As[m][k]  row-major, stride 40 -> A frag = 2x ds_load_b128
//   Bs[n][k]  transposed, stride 40 -> B frag = 2x ds_load_b128
// Epilogues:
//   0: C = v                         4: C = relu(v)^2
//   1: C = sigmoid(v + ex0[col])     5: C = sigmoid(v)
//   2: C = v + ex1[idx]              6: C += ex0[row]*ex1[idx]*v
//   3: C = tanh(v)                   7: C += ex0[row]*v
// ---------------------------------------------------------------------------
#define BM  128
#define BN  64
#define BKK 32
#define LDP 40   // padded K stride (shorts): 80B rows, 16B aligned, conflict-free

template<int EPI>
__global__ void __launch_bounds__(256) gemm_wmma(
    const float* __restrict__ A, const float* __restrict__ Bm, float* __restrict__ C,
    int M, int N, int K,
    const float* __restrict__ ex0, const float* __restrict__ ex1)
{
  __shared__ alignas(16) unsigned short As[2][BM][LDP];   // A[m][k]
  __shared__ alignas(16) unsigned short Bs[2][BN][LDP];   // B[n][k] (transposed)

  const int tid  = threadIdx.x;
  const int wid  = tid >> 5;
  const int lane = tid & 31;
  const int half = lane >> 4;
  const int l16  = lane & 15;
  const int mi   = wid & 3;
  const int ng   = wid >> 2;
  const int m0   = blockIdx.y * BM;
  const int n0   = blockIdx.x * BN;

  // Staging: A 128x32 -> 16 fp32/thread (one half-row); B 32x64 -> 8 fp32/thread.
  const int am = tid >> 1, ak = (tid & 1) << 4;   // A: row 0..127, col {0,16}
  const int bk = tid >> 3, bn = (tid & 7) << 3;   // B: k 0..31, n base 0..56

  v8f acc00 = {}, acc01 = {}, acc10 = {}, acc11 = {};

  const float* aRow = A  + (size_t)(m0 + am) * K + ak;
  const float* bRow = Bm + (size_t)bk * N + n0 + bn;
  const size_t bStep = (size_t)BKK * N;

  v8us aSt0, aSt1, bSt;                              // staged next tile (regs)
  // prologue: stage tile 0 and commit to buffer 0
  aSt0 = pack8(aRow);
  aSt1 = pack8(aRow + 8);
  bSt  = pack8(bRow);
  *(v8us*)&As[0][am][ak]     = aSt0;
  *(v8us*)&As[0][am][ak + 8] = aSt1;
  #pragma unroll
  for (int i = 0; i < 8; ++i) Bs[0][bn + i][bk] = bSt[i];

  const int nIter = K / BKK;
  for (int it = 0; it < nIter; ++it) {
    __syncthreads();                                 // buffer `cur` ready
    const int cur = it & 1;
    const bool more = (it + 1) < nIter;
    if (more) {                                      // overlap: load next tile
      const float* ap = aRow + (it + 1) * BKK;
      const float* bp = bRow + (size_t)(it + 1) * bStep;
      if (it + 2 < nIter) {                          // prefetch tile after next
        __builtin_prefetch(ap + BKK, 0, 3);
        __builtin_prefetch(bp + bStep, 0, 3);
      }
      aSt0 = pack8(ap);
      aSt1 = pack8(ap + 8);
      bSt  = pack8(bp);
    }

    // Fragments from current buffer: 2x b128 each.
    FragU a0, a1, b0, b1;
    const int ar0 = mi * 16 + l16;
    const int nc0 = ng * 32 + l16;
    const int ha  = half << 3;   // half*8
    const int hb  = half << 4;   // half*16
    a0.h[0] = *(const v8us*)&As[cur][ar0][ha];
    a0.h[1] = *(const v8us*)&As[cur][ar0][16 + ha];
    a1.h[0] = *(const v8us*)&As[cur][ar0 + 64][ha];
    a1.h[1] = *(const v8us*)&As[cur][ar0 + 64][16 + ha];
    b0.h[0] = *(const v8us*)&Bs[cur][nc0][ha];
    b0.h[1] = *(const v8us*)&Bs[cur][nc0][16 + ha];
    b1.h[0] = *(const v8us*)&Bs[cur][nc0 + 16][ha];
    b1.h[1] = *(const v8us*)&Bs[cur][nc0 + 16][16 + ha];

    acc00 = __builtin_amdgcn_wmma_f32_16x16x32_bf16(false, a0.b, false, b0.b,
                                                    (short)0, acc00, false, false);
    acc01 = __builtin_amdgcn_wmma_f32_16x16x32_bf16(false, a0.b, false, b1.b,
                                                    (short)0, acc01, false, false);
    acc10 = __builtin_amdgcn_wmma_f32_16x16x32_bf16(false, a1.b, false, b0.b,
                                                    (short)0, acc10, false, false);
    acc11 = __builtin_amdgcn_wmma_f32_16x16x32_bf16(false, a1.b, false, b1.b,
                                                    (short)0, acc11, false, false);

    if (more) {                                      // commit next tile
      const int nxt = cur ^ 1;
      *(v8us*)&As[nxt][am][ak]     = aSt0;
      *(v8us*)&As[nxt][am][ak + 8] = aSt1;
      #pragma unroll
      for (int i = 0; i < 8; ++i) Bs[nxt][bn + i][bk] = bSt[i];
    }
  }

  // C/D layout: VGPR i -> M = i + 8*half, N = lane%16
  const int rowA = m0 + mi * 16 + half * 8;
  const int colA = n0 + ng * 32 + l16;
  #pragma unroll
  for (int mt = 0; mt < 2; ++mt) {
    #pragma unroll
    for (int i = 0; i < 8; ++i) {
      const int row = rowA + mt * 64 + i;
      #pragma unroll
      for (int nt = 0; nt < 2; ++nt) {
        const int col = colA + nt * 16;
        float val;
        if (mt == 0) val = (nt == 0) ? acc00[i] : acc01[i];
        else         val = (nt == 0) ? acc10[i] : acc11[i];
        const size_t idx = (size_t)row * N + col;
        if constexpr (EPI == 0)      C[idx] = val;
        else if constexpr (EPI == 1) C[idx] = sigm(val + ex0[col]);
        else if constexpr (EPI == 2) C[idx] = val + ex1[idx];
        else if constexpr (EPI == 3) C[idx] = tanhf(val);
        else if constexpr (EPI == 4) { float r = fmaxf(val, 0.0f); C[idx] = r * r; }
        else if constexpr (EPI == 5) C[idx] = sigm(val);
        else if constexpr (EPI == 6) C[idx] += ex0[row] * ex1[idx] * val;
        else                         C[idx] += ex0[row] * val;
      }
    }
  }
}

// ---------------------------------------------------------------------------
// LayerNorm over last dim (C_), one block per token.
// ---------------------------------------------------------------------------
__global__ void __launch_bounds__(256) ln_kernel(
    const float* __restrict__ in, const float* __restrict__ w,
    const float* __restrict__ b, float* __restrict__ out)
{
  const int tok = blockIdx.x;
  const float* row = in + (size_t)tok * C_;
  __shared__ float red[256];
  float s = 0.f;
  for (int c = threadIdx.x; c < C_; c += 256) s += row[c];
  red[threadIdx.x] = s; __syncthreads();
  for (int o = 128; o > 0; o >>= 1) {
    if (threadIdx.x < o) red[threadIdx.x] += red[threadIdx.x + o];
    __syncthreads();
  }
  const float mean = red[0] * (1.0f / C_);
  __syncthreads();
  float vs = 0.f;
  for (int c = threadIdx.x; c < C_; c += 256) { float d = row[c] - mean; vs += d * d; }
  red[threadIdx.x] = vs; __syncthreads();
  for (int o = 128; o > 0; o >>= 1) {
    if (threadIdx.x < o) red[threadIdx.x] += red[threadIdx.x + o];
    __syncthreads();
  }
  const float rstd = rsqrtf(red[0] * (1.0f / C_) + 1e-5f);
  float* orow = out + (size_t)tok * C_;
  for (int c = threadIdx.x; c < C_; c += 256)
    orow[c] = (row[c] - mean) * rstd * w[c] + b[c];
}

// ---------------------------------------------------------------------------
// RWKV7 recurrence. One block per (b,h); 64 threads, each owning state
// column S[:, vd]; r/k/w broadcast via LDS.
//   S = S * w[:,None] + k[:,None] * v[None,:];  y = r @ S
// ---------------------------------------------------------------------------
__global__ void __launch_bounds__(64) rwkv_scan_kernel(
    const float* __restrict__ r, const float* __restrict__ k,
    const float* __restrict__ v, const float* __restrict__ w,
    float* __restrict__ y)
{
  const int b  = blockIdx.x / H_;
  const int hh = blockIdx.x % H_;
  const int t  = threadIdx.x;                 // vd index
  __shared__ float sr[HS_], sk[HS_], sw[HS_], sv[HS_];
  float S[HS_];
  #pragma unroll
  for (int i = 0; i < HS_; ++i) S[i] = 0.f;
  const size_t base = (size_t)b * T_ * C_ + (size_t)hh * HS_ + t;
  for (int step = 0; step < T_; ++step) {
    const size_t idx = base + (size_t)step * C_;
    sr[t] = r[idx]; sk[t] = k[idx]; sw[t] = w[idx]; sv[t] = v[idx];
    __syncthreads();
    const float vt = sv[t];
    float acc = 0.f;
    #pragma unroll
    for (int kd = 0; kd < HS_; ++kd) {
      S[kd] = S[kd] * sw[kd] + sk[kd] * vt;
      acc  += sr[kd] * S[kd];
    }
    y[idx] = acc;
    __syncthreads();
  }
}

// ---------------------------------------------------------------------------
// Market routing. One block per token; wave e of 8 computes the length-1024
// dots for expert e. gates layout: [E][BT].
// ---------------------------------------------------------------------------
__global__ void __launch_bounds__(256) routing_kernel(
    const float* __restrict__ h, const float* __restrict__ conf_w,
    const float* __restrict__ conf_b, const float* __restrict__ cap,
    const float* __restrict__ Wa, const float* __restrict__ ba,
    float* __restrict__ gates)
{
  const int tok  = blockIdx.x;
  const int e    = threadIdx.x >> 5;
  const int lane = threadIdx.x & 31;
  const float* hr = h + (size_t)tok * C_;
  float dc = 0.f, da = 0.f;
  for (int c = lane; c < C_; c += 32) {
    const float hv = hr[c];
    dc += hv * conf_w[(size_t)e * C_ + c];   // conf_w: (E,C)
    da += hv * Wa[(size_t)c * E_ + e];       // Wa: (C,E)
  }
  #pragma unroll
  for (int o = 16; o > 0; o >>= 1) { dc += __shfl_down(dc, o); da += __shfl_down(da, o); }
  __shared__ float sconf[E_], saff[E_];
  if (lane == 0) { sconf[e] = dc; saff[e] = da; }
  __syncthreads();
  if (threadIdx.x == 0) {
    float conf[E_], aff[E_];
    float m = -1e30f;
    for (int i = 0; i < E_; ++i) {
      conf[i] = sigm(sconf[i] + conf_b[i]);
      aff[i]  = saff[i] + ba[i];
      m = fmaxf(m, aff[i]);
    }
    float ssum = 0.f;
    for (int i = 0; i < E_; ++i) { aff[i] = __expf(aff[i] - m); ssum += aff[i]; }
    float bids[E_];
    for (int i = 0; i < E_; ++i) bids[i] = conf[i] * cap[i] + aff[i] / ssum;
    int b0i = 0; float b0v = bids[0];
    for (int i = 1; i < E_; ++i) if (bids[i] > b0v) { b0v = bids[i]; b0i = i; }
    int b1i = -1; float b1v = -1e30f;
    for (int i = 0; i < E_; ++i) if (i != b0i && bids[i] > b1v) { b1v = bids[i]; b1i = i; }
    const float w0 = 1.0f / (1.0f + __expf(b1v - b0v));
    const float w1 = 1.0f - w0;
    for (int i = 0; i < E_; ++i)
      gates[(size_t)i * BT_ + tok] = (i == b0i) ? w0 : ((i == b1i) ? w1 : 0.0f);
  }
}

// Concatenate [h | y] -> hcat (BT, 2C)
__global__ void __launch_bounds__(256) concat_kernel(
    const float* __restrict__ h, const float* __restrict__ y, float* __restrict__ hc)
{
  const size_t i   = (size_t)blockIdx.x * blockDim.x + threadIdx.x;
  const size_t tok = i / (2 * C_);
  const int    j   = (int)(i % (2 * C_));
  hc[i] = (j < C_) ? h[tok * C_ + j] : y[tok * C_ + (j - C_)];
}

// Cross-attention of one token over its P=4 bridge-prefix slots, per head.
// One thread per (token, head).
__global__ void __launch_bounds__(256) prefix_attn_kernel(
    const float* __restrict__ q, const float* __restrict__ kp,
    const float* __restrict__ vp, float* __restrict__ o)
{
  const int gid = blockIdx.x * blockDim.x + threadIdx.x;
  const int tok = gid >> 4;
  const int hh  = gid & (H_ - 1);
  const float* qp = q + (size_t)tok * C_ + hh * HS_;
  float sc[P_];
  #pragma unroll
  for (int p = 0; p < P_; ++p) {
    const float* kpp = kp + ((size_t)tok * P_ + p) * C_ + hh * HS_;
    float s = 0.f;
    for (int d = 0; d < HS_; ++d) s += qp[d] * kpp[d];
    sc[p] = s * 0.125f;                       // 1/sqrt(64)
  }
  float m = sc[0];
  #pragma unroll
  for (int p = 1; p < P_; ++p) m = fmaxf(m, sc[p]);
  float ssum = 0.f;
  #pragma unroll
  for (int p = 0; p < P_; ++p) { sc[p] = __expf(sc[p] - m); ssum += sc[p]; }
  const float inv = 1.0f / ssum;
  float* op = o + (size_t)tok * C_ + hh * HS_;
  for (int d = 0; d < HS_; ++d) {
    float acc = 0.f;
    #pragma unroll
    for (int p = 0; p < P_; ++p)
      acc += sc[p] * vp[((size_t)tok * P_ + p) * C_ + hh * HS_ + d];
    op[d] = acc * inv;
  }
}

// ---------------------------------------------------------------------------
extern "C" void kernel_launch(void* const* d_in, const int* in_sizes, int n_in,
                              void* d_out, int out_size, void* d_ws, size_t ws_size,
                              hipStream_t stream) {
  (void)in_sizes; (void)n_in; (void)out_size; (void)ws_size;
  const float* x     = (const float*)d_in[0];
  const float* cap   = (const float*)d_in[2];
  const float* ln1w  = (const float*)d_in[3];
  const float* ln1b  = (const float*)d_in[4];
  const float* ln2w  = (const float*)d_in[5];
  const float* ln2b  = (const float*)d_in[6];
  const float* Wr    = (const float*)d_in[7];
  const float* Wk    = (const float*)d_in[8];
  const float* Wv    = (const float*)d_in[9];
  const float* Ww    = (const float*)d_in[10];
  const float* wbias = (const float*)d_in[11];
  const float* Wo    = (const float*)d_in[12];
  const float* confw = (const float*)d_in[13];
  const float* confb = (const float*)d_in[14];
  const float* Wa    = (const float*)d_in[17];
  const float* ba    = (const float*)d_in[18];
  const float* Wb    = (const float*)d_in[19];
  const float* ffnWr = (const float*)d_in[20];
  const float* ffnWk = (const float*)d_in[21];
  const float* ffnWv = (const float*)d_in[22];
  const float* trWq  = (const float*)d_in[23];
  const float* trWk  = (const float*)d_in[24];
  const float* trWv  = (const float*)d_in[25];
  const float* trWo  = (const float*)d_in[26];

  float* out   = (float*)d_out;
  float* xout  = out;                          // x_after_att then += expert mix
  float* vflat = out + (size_t)BT_ * C_;       // second tuple output

  // Workspace layout (floats); regions reused across phases.
  const size_t M4 = (size_t)BT_ * C_;          // 4M floats
  float* ws     = (float*)d_ws;
  float* xln    = ws;                          // 1x M4 ; later: attn_out
  float* rbuf   = ws + 1 * M4;                 // 1x M4 ; later: sig_r / q
  float* kbuf   = ws + 2 * M4;                 // 1x M4
  float* wbuf   = ws + 3 * M4;                 // 1x M4 ; later: gates (E*BT)
  float* ybuf   = ws + 4 * M4;                 // rwkv_state
  float* hbuf   = ws + 5 * M4;                 // ln2 output
  float* hcat   = ws + 6 * M4;                 // 2x M4
  float* prefix = ws + 8 * M4;                 // 4x M4 (BT x P*C)
  float* tmp1   = ws + 12 * M4;                // 4x M4 (kk / kp)
  float* tmp3   = ws + 16 * M4;                // 4x M4 (vp)
  float* gates  = wbuf;
  float* tmp2   = rbuf;
  float* attnO  = xln;

  const dim3 blk(256);
  const dim3 gP(C_ / BN, BT_ / BM);            // (16, 32): BTxC GEMMs
  const dim3 gBr(FF_ / BN, BT_ / BM);          // (64, 32): bridge / ffn-k
  const dim3 gPr(C_ / BN, (BT_ * P_) / BM);    // (16, 128): prefix projections

  // 1) ln1
  ln_kernel<<<BT_, blk, 0, stream>>>(x, ln1w, ln1b, xln);
  // 2) r/k/v/w projections (w gets sigmoid + bias epilogue)
  gemm_wmma<0><<<gP, blk, 0, stream>>>(xln, Wr, rbuf,  BT_, C_, C_, nullptr, nullptr);
  gemm_wmma<0><<<gP, blk, 0, stream>>>(xln, Wk, kbuf,  BT_, C_, C_, nullptr, nullptr);
  gemm_wmma<0><<<gP, blk, 0, stream>>>(xln, Wv, vflat, BT_, C_, C_, nullptr, nullptr);
  gemm_wmma<1><<<gP, blk, 0, stream>>>(xln, Ww, wbuf,  BT_, C_, C_, wbias,   nullptr);
  // 3) sequential RWKV scan
  rwkv_scan_kernel<<<B_ * H_, 64, 0, stream>>>(rbuf, kbuf, vflat, wbuf, ybuf);
  // 4) output projection + residual -> x_after_att (in d_out)
  gemm_wmma<2><<<gP, blk, 0, stream>>>(ybuf, Wo, xout, BT_, C_, C_, nullptr, x);
  // 5) ln2
  ln_kernel<<<BT_, blk, 0, stream>>>(xout, ln2w, ln2b, hbuf);
  // 6) routing -> gates[E][BT]   (scan done, wbuf free)
  routing_kernel<<<BT_, blk, 0, stream>>>(hbuf, confw, confb, cap, Wa, ba, gates);
  // 7) bridge prefix = tanh([h|state] @ Wb)
  concat_kernel<<<(unsigned)((2 * M4) / 256), blk, 0, stream>>>(hbuf, ybuf, hcat);
  gemm_wmma<3><<<gBr, blk, 0, stream>>>(hcat, Wb, prefix, BT_, FF_, 2 * C_, nullptr, nullptr);
  // 8) RWKV-FFN experts (dense, gated accumulate into x_out)
  for (int e = 0; e < NR_; ++e) {
    gemm_wmma<4><<<gBr, blk, 0, stream>>>(hbuf, ffnWk + (size_t)e * C_ * FF_, tmp1,
                                          BT_, FF_, C_, nullptr, nullptr);
    gemm_wmma<5><<<gP,  blk, 0, stream>>>(hbuf, ffnWr + (size_t)e * C_ * C_,  tmp2,
                                          BT_, C_,  C_, nullptr, nullptr);
    gemm_wmma<6><<<gP,  blk, 0, stream>>>(tmp1, ffnWv + (size_t)e * FF_ * C_, xout,
                                          BT_, C_, FF_, gates + (size_t)e * BT_, tmp2);
  }
  // 9) transformer experts (gated accumulate into x_out)
  for (int t = 0; t < NT_; ++t) {
    const int e = NR_ + t;
    gemm_wmma<0><<<gP,  blk, 0, stream>>>(hbuf,   trWq + (size_t)t * C_ * C_, tmp2,
                                          BT_, C_, C_, nullptr, nullptr);
    gemm_wmma<0><<<gPr, blk, 0, stream>>>(prefix, trWk + (size_t)t * C_ * C_, tmp1,
                                          BT_ * P_, C_, C_, nullptr, nullptr);
    gemm_wmma<0><<<gPr, blk, 0, stream>>>(prefix, trWv + (size_t)t * C_ * C_, tmp3,
                                          BT_ * P_, C_, C_, nullptr, nullptr);
    prefix_attn_kernel<<<(BT_ * H_) / 256, blk, 0, stream>>>(tmp2, tmp1, tmp3, attnO);
    gemm_wmma<7><<<gP,  blk, 0, stream>>>(attnO, trWo + (size_t)t * C_ * C_, xout,
                                          BT_, C_, C_, gates + (size_t)e * BT_, nullptr);
  }
}
